// DeepAir_40123584479294
// MI455X (gfx1250) — compile-verified, matrix operand order checked by reading, not run on hardware
//
#include <hip/hip_runtime.h>
#include <hip/hip_bf16.h>
#include <math.h>

typedef __attribute__((ext_vector_type(16))) _Float16 v16h;
typedef __attribute__((ext_vector_type(8)))  float    v8f;

#define BB 16
#define TT 24
#define NN 300
#define EE 9600
#define HH 3
#define FF 8
#define HF 24
#define GH 16
#define OUTW 7200   // NUM_AIRPORTS * HORIZON

// ---------- order-preserving float <-> uint map for atomic max ----------
__device__ __forceinline__ unsigned encf(float x) {
    unsigned u = __float_as_uint(x);
    return (u & 0x80000000u) ? ~u : (u | 0x80000000u);
}
__device__ __forceinline__ float decf(unsigned u) {
    return __uint_as_float((u & 0x80000000u) ? (u & 0x7FFFFFFFu) : ~u);
}
// single-instruction v_rcp_f32 instead of IEEE div chains
__device__ __forceinline__ float fast_rcp(float x) { return __builtin_amdgcn_rcpf(x); }
__device__ __forceinline__ float sigmf(float x) { return fast_rcp(1.0f + __expf(-x)); }
__device__ __forceinline__ float tanh_fast(float x) {
    // tanh(x) = 1 - 2/(exp(2x)+1); exp overflow/underflow saturate to +-1 correctly
    return 1.0f - 2.0f * fast_rcp(__expf(2.0f * x) + 1.0f);
}

// =====================================================================
// Kernel A: per-(b,t) EdgeGAT + edge softmax + pooled reduction.
// gru_in[b][t][k] = W_node[k]*meanS[h(k)] + gat_bias[k]; the rank-1
// projections collapse the GAT to per-head scalar logits.
// LDS atomics: ds_max_u32 (seg-max), ds_add_f32 (seg-sum / reduce).
// =====================================================================
__global__ void __launch_bounds__(256)
gat_pool_kernel(const float* __restrict__ x, const float* __restrict__ ew,
                const float* __restrict__ W_node, const float* __restrict__ W_edge,
                const float* __restrict__ attn_l, const float* __restrict__ attn_r,
                const float* __restrict__ attn_e, const float* __restrict__ gat_bias,
                const int* __restrict__ src, const int* __restrict__ dst,
                float* __restrict__ gru_in)
{
    __shared__ float    xs[NN];
    __shared__ unsigned menc[HH * NN];
    __shared__ float    den[HH * NN];
    __shared__ float    redS[HH];

    const int bt  = blockIdx.x;        // 0..B*T-1
    const int tid = threadIdx.x;
    const size_t nbase = (size_t)bt * NN;
    const size_t ebase = (size_t)bt * EE;

    // head coefficients (rank-1 collapse of the GAT projections)
    float cl[HH], cr[HH], ce[HH];
#pragma unroll
    for (int h = 0; h < HH; ++h) {
        float a = 0.f, b = 0.f, c = 0.f;
#pragma unroll
        for (int f = 0; f < FF; ++f) {
            a += W_node[h * FF + f] * attn_l[h * FF + f];
            b += W_node[h * FF + f] * attn_r[h * FF + f];
            c += W_edge[h * FF + f] * attn_e[h * FF + f];
        }
        cl[h] = a; cr[h] = b; ce[h] = c;
    }

    for (int i = tid; i < NN; i += 256) xs[i] = x[nbase + i];
    for (int i = tid; i < HH * NN; i += 256) { menc[i] = 0u; den[i] = 0.f; }
    if (tid < HH) redS[tid] = 0.f;
    __syncthreads();

    // pass 1: segment max over incoming edges of each dst
    for (int e = tid; e < EE; e += 256) {
        __builtin_prefetch(&ew[ebase + e + 512], 0, 0);   // global_prefetch_b8
        float w = ew[ebase + e];
        int s = src[e], d = dst[e];
        float xsv = xs[s], xdv = xs[d];
#pragma unroll
        for (int h = 0; h < HH; ++h) {
            float v = xsv * cl[h] + xdv * cr[h] + w * ce[h];
            v = (v > 0.f) ? v : 0.2f * v;
            atomicMax(&menc[h * NN + d], encf(v));
        }
    }
    __syncthreads();

    // pass 2: segment sum of exp(e - max)
    for (int e = tid; e < EE; e += 256) {
        float w = ew[ebase + e];
        int s = src[e], d = dst[e];
        float xsv = xs[s], xdv = xs[d];
#pragma unroll
        for (int h = 0; h < HH; ++h) {
            float v = xsv * cl[h] + xdv * cr[h] + w * ce[h];
            v = (v > 0.f) ? v : 0.2f * v;
            float num = __expf(v - decf(menc[h * NN + d]));
            atomicAdd(&den[h * NN + d], num);
        }
    }
    __syncthreads();

    // pass 3: Σ_e alpha_e * x_src   (pooled message, per head)
    float acc[HH] = {0.f, 0.f, 0.f};
    for (int e = tid; e < EE; e += 256) {
        float w = ew[ebase + e];
        int s = src[e], d = dst[e];
        float xsv = xs[s], xdv = xs[d];
#pragma unroll
        for (int h = 0; h < HH; ++h) {
            float v = xsv * cl[h] + xdv * cr[h] + w * ce[h];
            v = (v > 0.f) ? v : 0.2f * v;
            float num = __expf(v - decf(menc[h * NN + d]));
            acc[h] += num * fast_rcp(den[h * NN + d]) * xsv;
        }
    }
#pragma unroll
    for (int h = 0; h < HH; ++h) atomicAdd(&redS[h], acc[h]);
    __syncthreads();

    if (tid < HF) {
        float v = W_node[tid] * (redS[tid / FF] * (1.0f / NN)) + gat_bias[tid];
        gru_in[(size_t)bt * HF + tid] = v;   // [b][t][24] since bt = b*T + t
    }
}

// ---------- WMMA operand builders (f16 16x16x32) ----------
// Branchless: clamp address, select value -> no exec-mask divergence.
// B operand: B[k,n] = W[rbase+n, k]  (W row-major with leading dim ld)
__device__ __forceinline__ v16h load_B(const float* __restrict__ W, int rbase,
                                       int ld, int kvalid, int lane)
{
    int n   = lane & 15;
    int kb  = (lane < 16) ? 0 : 8;
    int kb2 = (lane < 16) ? 16 : 24;
    const float* row = W + (size_t)(rbase + n) * ld;
    v16h b;
#pragma unroll
    for (int i = 0; i < 8; ++i) {
        int k = kb + i;
        float v = row[(k < kvalid) ? k : 0];
        b[i] = (k < kvalid) ? (_Float16)v : (_Float16)0.f;
    }
#pragma unroll
    for (int i = 0; i < 8; ++i) {
        int k = kb2 + i;
        float v = row[(k < kvalid) ? k : 0];
        b[8 + i] = (k < kvalid) ? (_Float16)v : (_Float16)0.f;
    }
    return b;
}
// A operand: A[m,k] = R[m*ld + k]
__device__ __forceinline__ v16h load_A(const float* __restrict__ R, int ld,
                                       int kvalid, int lane)
{
    int m   = lane & 15;
    int kb  = (lane < 16) ? 0 : 8;
    int kb2 = (lane < 16) ? 16 : 24;
    const float* row = R + (size_t)m * ld;
    v16h a;
#pragma unroll
    for (int i = 0; i < 8; ++i) {
        int k = kb + i;
        float v = row[(k < kvalid) ? k : 0];
        a[i] = (k < kvalid) ? (_Float16)v : (_Float16)0.f;
    }
#pragma unroll
    for (int i = 0; i < 8; ++i) {
        int k = kb2 + i;
        float v = row[(k < kvalid) ? k : 0];
        a[8 + i] = (k < kvalid) ? (_Float16)v : (_Float16)0.f;
    }
    return a;
}
__device__ __forceinline__ v8f wmma16(v16h a, v16h b, v8f c) {
    return __builtin_amdgcn_wmma_f32_16x16x32_f16(false, a, false, b, (short)0, c,
                                                  false, false);
}

// =====================================================================
// Kernel B: GRU over T=24 steps, one wave. Six v_wmma per step:
// gi = xt @ W_ih^T (3 tiles, K=24), gh = h @ W_hh^T (3 tiles, K=16).
// Hidden state cycled through LDS (C-layout -> A-layout).
// =====================================================================
__global__ void __launch_bounds__(32)
gru_kernel(const float* __restrict__ gru_in, const float* __restrict__ W_ih,
           const float* __restrict__ W_hh, const float* __restrict__ b_ih,
           const float* __restrict__ b_hh, float* __restrict__ h_out)
{
    __shared__ float hbuf[GH][GH];
    const int lane = threadIdx.x;

    // invariant B operands: gate tiles r/z/n at row offsets 0/16/32
    v16h Bi[3], Bh[3];
    float bi[3], bh[3];
#pragma unroll
    for (int t16 = 0; t16 < 3; ++t16) {
        Bi[t16] = load_B(W_ih, t16 * 16, HF, HF, lane);
        Bh[t16] = load_B(W_hh, t16 * 16, GH, GH, lane);
        bi[t16] = b_ih[t16 * 16 + (lane & 15)];
        bh[t16] = b_hh[t16 * 16 + (lane & 15)];
    }

    // zero hidden state (LDS, row-major 16x16)
#pragma unroll
    for (int i = 0; i < 8; ++i) {
        int idx = lane * 8 + i;
        hbuf[idx >> 4][idx & 15] = 0.f;
    }
    // C-layout copy of h held per lane: element i -> (m = i + 8*(lane>=16), j = lane&15)
    float hc[8];
#pragma unroll
    for (int i = 0; i < 8; ++i) hc[i] = 0.f;
    __syncthreads();

    const int m  = lane & 15;      // A-operand row for this lane
    const int hi = lane >> 4;      // half-wave select
    for (int t = 0; t < TT; ++t) {
        // xt A operand: xt[m,k] = gru_in[(m*T + t)*24 + k], K=24 padded to 32
        v16h ax = load_A(gru_in + (size_t)t * HF, TT * HF, HF, lane);
        // h A operand from LDS, K=16 padded
        v16h ah;
        {
            int kb = hi ? 8 : 0;
#pragma unroll
            for (int i = 0; i < 8; ++i) ah[i] = (_Float16)hbuf[m][kb + i];
#pragma unroll
            for (int i = 0; i < 8; ++i) ah[8 + i] = (_Float16)0.f;
        }
        v8f z8 = {};
        v8f gi0 = wmma16(ax, Bi[0], z8), gi1 = wmma16(ax, Bi[1], z8), gi2 = wmma16(ax, Bi[2], z8);
        v8f gh0 = wmma16(ah, Bh[0], z8), gh1 = wmma16(ah, Bh[1], z8), gh2 = wmma16(ah, Bh[2], z8);
        __syncthreads();           // done reading hbuf
#pragma unroll
        for (int i = 0; i < 8; ++i) {
            float r = sigmf(gi0[i] + bi[0] + gh0[i] + bh[0]);
            float z = sigmf(gi1[i] + bi[1] + gh1[i] + bh[1]);
            float n = tanh_fast(gi2[i] + bi[2] + r * (gh2[i] + bh[2]));
            hc[i] = (1.0f - z) * n + z * hc[i];
            hbuf[i + 8 * hi][m] = hc[i];   // C-layout -> row-major
        }
        __syncthreads();
    }
    // write h_n [16,16] row-major to workspace
#pragma unroll
    for (int i = 0; i < 8; ++i) h_out[(i + 8 * hi) * GH + m] = hc[i];
}

// =====================================================================
// Kernel C: out[16,7200] = h_n @ W_fc^T + b_fc. 450 tiles, 1 wave/tile,
// one v_wmma_f32_16x16x32_f16 per tile.
// =====================================================================
__global__ void __launch_bounds__(32)
fc_kernel(const float* __restrict__ h_n, const float* __restrict__ W_fc,
          const float* __restrict__ b_fc, float* __restrict__ out)
{
    const int lane  = threadIdx.x;
    const int obase = blockIdx.x * 16;

    v16h a = load_A(h_n, GH, GH, lane);            // h_n as A, K=16 padded
    v16h b = load_B(W_fc, obase, GH, GH, lane);    // W_fc rows as B columns
    v8f  c = {};
    c = wmma16(a, b, c);

    const int   col  = obase + (lane & 15);
    const float bia  = b_fc[col];
    const int   mofs = 8 * (lane >> 4);
#pragma unroll
    for (int i = 0; i < 8; ++i)
        out[(size_t)(i + mofs) * OUTW + col] = c[i] + bia;
}

extern "C" void kernel_launch(void* const* d_in, const int* in_sizes, int n_in,
                              void* d_out, int out_size, void* d_ws, size_t ws_size,
                              hipStream_t stream)
{
    const float* x        = (const float*)d_in[0];
    const float* ew       = (const float*)d_in[1];
    const float* W_node   = (const float*)d_in[2];
    const float* W_edge   = (const float*)d_in[3];
    const float* attn_l   = (const float*)d_in[4];
    const float* attn_r   = (const float*)d_in[5];
    const float* attn_e   = (const float*)d_in[6];
    const float* gat_bias = (const float*)d_in[7];
    const float* W_ih     = (const float*)d_in[8];
    const float* W_hh     = (const float*)d_in[9];
    const float* b_ih     = (const float*)d_in[10];
    const float* b_hh     = (const float*)d_in[11];
    const float* W_fc     = (const float*)d_in[12];
    const float* b_fc     = (const float*)d_in[13];
    const int*   src      = (const int*)d_in[14];
    const int*   dst      = (const int*)d_in[15];

    float* ws     = (float*)d_ws;
    float* gru_in = ws;                        // B*T*24 = 9216 floats
    float* h_ws   = ws + (size_t)BB * TT * HF; // 256 floats

    gat_pool_kernel<<<dim3(BB * TT), 256, 0, stream>>>(
        x, ew, W_node, W_edge, attn_l, attn_r, attn_e, gat_bias, src, dst, gru_in);
    gru_kernel<<<1, 32, 0, stream>>>(gru_in, W_ih, W_hh, b_ih, b_hh, h_ws);
    fc_kernel<<<OUTW / 16, 32, 0, stream>>>(h_ws, W_fc, b_fc, (float*)d_out);
}
